// PolyVAE_repara_36825049596473
// MI455X (gfx1250) — matrix operand order repaired and checked by reading, not compiled, not run on hardware
//
#include <hip/hip_runtime.h>
#include <hip/hip_fp16.h>
#include <hip/hip_bf16.h>

// ---------------------------------------------------------------------------
// PolyVAE forward for MI455X (gfx1250): wave32 + v_wmma_f32_16x16x32_f16.
// Sequential GRU scans run as persistent single-workgroup kernels (32 waves,
// LDS-resident double-buffered hidden state); parallel projections use a
// generic WMMA GEMM. Weights converted once per call to f16.
// ---------------------------------------------------------------------------

typedef __attribute__((ext_vector_type(16))) _Float16 v16h;
typedef __attribute__((ext_vector_type(8)))  float    v8f;

#define NB   64     // batch
#define NT   320    // seq len
#define NH   512    // hidden
#define NG   1536   // 3*H
#define NZ   1024   // latent
#define NE   10     // embed dim
#define NV   90     // vocab
#define NBEAT 20
#define NTICK 16

__device__ __forceinline__ v8f wmma_(v16h a, v16h b, v8f c) {
  return __builtin_amdgcn_wmma_f32_16x16x32_f16(
      false, a, false, b, (short)0, c, false, false);
}

// A fragment: 16x32 f16 tile of row-major matrix A (lda halfs per row).
// lanes 0-15: row m0+lane, K {k0..k0+7, k0+16..k0+23}
// lanes 16-31: row m0+lane-16, K {k0+8..k0+15, k0+24..k0+31}
__device__ __forceinline__ v16h load_frag_a(const _Float16* A, int lda,
                                            int m0, int k0, int lane) {
  const _Float16* p = A + (size_t)(m0 + (lane & 15)) * lda + k0 + ((lane & 16) ? 8 : 0);
  union { v16h v; float4 f[2]; } u;
  u.f[0] = *(const float4*)p;
  u.f[1] = *(const float4*)(p + 16);
  return u.v;
}

// B fragment: 32x16 (KxN) from W stored row-major as N x K (ldw halfs/row);
// B[k][n] = W[n][k]. lane n = n0+lane%16; K same order as A fragment.
__device__ __forceinline__ v16h load_frag_b(const _Float16* W, int ldw,
                                            int n0, int k0, int lane) {
  const _Float16* p = W + (size_t)(n0 + (lane & 15)) * ldw + k0 + ((lane & 16) ? 8 : 0);
  union { v16h v; float4 f[2]; } u;
  u.f[0] = *(const float4*)p;
  u.f[1] = *(const float4*)(p + 16);
  return u.v;
}

__device__ __forceinline__ float sigmoidf_(float x) { return 1.f / (1.f + __expf(-x)); }
__device__ __forceinline__ float seluf_(float x) {
  const float a = 1.6732632423543772f, s = 1.0507009873554805f;
  return x > 0.f ? s * x : s * a * (__expf(x) - 1.f);
}

// ---------------------------------------------------------------------------
// f32 -> f16 conversion / repack / zero-pad
// ---------------------------------------------------------------------------
__global__ void cvt_f16(const float* __restrict__ src, _Float16* __restrict__ dst,
                        int rows, int cols, int sstride, int soff, int dstride,
                        int srows) {
  int i = blockIdx.x * blockDim.x + threadIdx.x;
  if (i >= rows * cols) return;
  int r = i / cols, c = i % cols;
  float v = (r < srows) ? src[(size_t)r * sstride + soff + c] : 0.f;
  dst[(size_t)r * dstride + c] = (_Float16)v;
}

// constant beat input projection: gib0[n] = beat0 * Wih_b0[n] + bih_b0[n]
__global__ void gib0_k(const float* beat0, const float* wih, const float* bih,
                       float* out) {
  int i = blockIdx.x * blockDim.x + threadIdx.x;
  if (i < NG) out[i] = beat0[0] * wih[i] + bih[i];
}

// var = exp(lv); z = mu + var*std_z  (faithful to reference: var scaling)
__global__ void reparam_k(const float* __restrict__ mu, const float* __restrict__ lv,
                          const float* __restrict__ stdz, float* __restrict__ zout,
                          float* __restrict__ varout, _Float16* __restrict__ z16,
                          int n) {
  int i = blockIdx.x * blockDim.x + threadIdx.x;
  if (i >= n) return;
  float v = expf(lv[i]);
  float z = mu[i] + v * stdz[i];
  varout[i] = v;
  zout[i] = z;
  z16[i] = (_Float16)z;
}

// ---------------------------------------------------------------------------
// Generic WMMA GEMM: C(MxN,f32/f16) = act(A(MxK,f16) @ W(NxK,f16)^T + bias)
// One 16x16 tile per wave, 8 waves / block. act: 0=none 1=selu 2=relu
// ---------------------------------------------------------------------------
__global__ __launch_bounds__(256) void gemm_wmma(
    const _Float16* __restrict__ A, int lda, const _Float16* __restrict__ W,
    int ldw, const float* __restrict__ bias, float* __restrict__ Cf,
    _Float16* __restrict__ Ch, int ldc, int M, int N, int K, int act) {
  int wave = threadIdx.x >> 5, lane = threadIdx.x & 31;
  int ntiles = N >> 4, mtiles = M >> 4;
  int idx = blockIdx.x * 8 + wave;
  if (idx >= ntiles * mtiles) return;
  int m0 = (idx / ntiles) * 16, n0 = (idx % ntiles) * 16;
  v8f acc = {};
  for (int k0 = 0; k0 < K; k0 += 32)
    acc = wmma_(load_frag_a(A, lda, m0, k0, lane),
                load_frag_b(W, ldw, n0, k0, lane), acc);
  int col = n0 + (lane & 15);
  int rb = m0 + ((lane & 16) ? 8 : 0);
  float bv = bias ? bias[col] : 0.f;
  for (int e = 0; e < 8; ++e) {
    float v = acc[e] + bv;
    if (act == 1) v = seluf_(v);
    else if (act == 2) v = fmaxf(v, 0.f);
    if (Cf) Cf[(size_t)(rb + e) * ldc + col] = v;
    if (Ch) Ch[(size_t)(rb + e) * ldc + col] = (_Float16)v;
  }
}

// ---------------------------------------------------------------------------
// Persistent GRU scan. One workgroup (32 waves) per sequence/direction.
// h state: f16, double-buffered in LDS. Per step each wave owns 4 (m,j)
// 16x16 output tiles and 3-4 WMMA accumulators (gates r/z/n).
//   mode 0: input = embedding[x[b,t]] (E=10) via VALU dots   (encoder L0)
//   mode 1: input = xs_f16(T,B,Din) fused WMMA with Wih      (encoder L1, beat L1)
//   mode 2: input projection constant vector givec[1536]     (beat L0)
// ---------------------------------------------------------------------------
struct GruJob {
  const _Float16* Whh;   // NG x NH f16
  const float* bhh;      // NG
  const float* bih;      // NG (modes 0,1)
  const float* h0;       // B x h0_stride f32 or null
  int h0_stride;
  int mode;
  // mode 0
  const int* xtok;       // B x T
  const float* emb;      // V x E f32
  const float* WihF32;   // NG x E f32
  int E_;
  // mode 1
  const _Float16* xs;    // T x B x Din f16
  const _Float16* Wih;   // NG x Din f16
  int Din;
  // mode 2
  const float* givec;    // NG (includes bih)
  // outputs
  _Float16* hout_f16;    // (T*B) x out_stride (col-offset pre-applied) or null
  int out_stride;
  _Float16* hfinal_f16;  // B x fin_stride (col-offset pre-applied) or null
  int fin_stride;
  int T;
  int reverse;
};

__global__ __launch_bounds__(1024) void gru_seq(GruJob ja, GruJob jb) {
  GruJob j = (blockIdx.x == 0) ? ja : jb;
  extern __shared__ char smem[];
  _Float16* hbuf = (_Float16*)smem;                 // 2 * 64*512 f16
  float* xemb = (float*)(hbuf + 2 * NB * NH);       // 64 x 16 f32
  const int tid = threadIdx.x;
  const int lane = tid & 31;
  const int wave = tid >> 5;

  for (int i = tid; i < NB * NH; i += 1024) {
    int b = i >> 9, c = i & (NH - 1);
    float hv = j.h0 ? j.h0[(size_t)b * j.h0_stride + c] : 0.f;
    hbuf[i] = (_Float16)hv;
  }
  int cur = 0;
  for (int s = 0; s < j.T; ++s) {
    int t = j.reverse ? (j.T - 1 - s) : s;
    __syncthreads();
    if (j.mode == 0) {
      for (int i = tid; i < NB * NE; i += 1024) {
        int b = i / NE, e = i % NE;
        int tok = j.xtok[b * j.T + t];
        xemb[b * 16 + e] = j.emb[tok * NE + e];
      }
    }
    __syncthreads();
    _Float16* hcur = hbuf + cur * (NB * NH);
    _Float16* hnxt = hbuf + (cur ^ 1) * (NB * NH);

    for (int p = 0; p < 4; ++p) {
      int pid = wave * 4 + p;
      int m0 = (pid >> 5) * 16, n0 = (pid & 31) * 16;
      v8f ar = {}, az = {}, anh = {}, ani = {};
      for (int k0 = 0; k0 < NH; k0 += 32) {
        v16h a = load_frag_a(hcur, NH, m0, k0, lane);
        ar  = wmma_(a, load_frag_b(j.Whh, NH, n0,            k0, lane), ar);
        az  = wmma_(a, load_frag_b(j.Whh, NH, n0 + NH,       k0, lane), az);
        anh = wmma_(a, load_frag_b(j.Whh, NH, n0 + 2 * NH,   k0, lane), anh);
      }
      if (j.mode == 1) {
        const _Float16* xa = j.xs + (size_t)t * NB * j.Din;
        for (int k0 = 0; k0 < j.Din; k0 += 32) {
          v16h a = load_frag_a(xa, j.Din, m0, k0, lane);
          ar  = wmma_(a, load_frag_b(j.Wih, j.Din, n0,          k0, lane), ar);
          az  = wmma_(a, load_frag_b(j.Wih, j.Din, n0 + NH,     k0, lane), az);
          ani = wmma_(a, load_frag_b(j.Wih, j.Din, n0 + 2 * NH, k0, lane), ani);
        }
      }
      int colj = n0 + (lane & 15);
      int rb = m0 + ((lane & 16) ? 8 : 0);
      float bhr = j.bhh[colj], bhz = j.bhh[NH + colj], bhn = j.bhh[2 * NH + colj];
      for (int e = 0; e < 8; ++e) {
        int b = rb + e;
        float gr, gz, gn;
        if (j.mode == 0) {
          float sr = j.bih[colj], sz = j.bih[NH + colj], sn = j.bih[2 * NH + colj];
          for (int ee = 0; ee < j.E_; ++ee) {
            float xv = xemb[b * 16 + ee];
            sr += xv * j.WihF32[(size_t)colj * j.E_ + ee];
            sz += xv * j.WihF32[(size_t)(NH + colj) * j.E_ + ee];
            sn += xv * j.WihF32[(size_t)(2 * NH + colj) * j.E_ + ee];
          }
          gr = sr; gz = sz; gn = sn;
        } else if (j.mode == 1) {
          gr = j.bih[colj];
          gz = j.bih[NH + colj];
          gn = ani[e] + j.bih[2 * NH + colj];
        } else {
          gr = j.givec[colj];
          gz = j.givec[NH + colj];
          gn = j.givec[2 * NH + colj];
        }
        float r = sigmoidf_(gr + ar[e] + bhr);
        float z = sigmoidf_(gz + az[e] + bhz);
        float n = tanhf(gn + r * (anh[e] + bhn));
        float ho = (float)hcur[b * NH + colj];
        float hn = (1.f - z) * n + z * ho;
        hnxt[b * NH + colj] = (_Float16)hn;
        if (j.hout_f16)
          j.hout_f16[((size_t)t * NB + b) * j.out_stride + colj] = (_Float16)hn;
        if (s == j.T - 1 && j.hfinal_f16)
          j.hfinal_f16[(size_t)b * j.fin_stride + colj] = (_Float16)hn;
      }
    }
    cur ^= 1;
  }
}

// ---------------------------------------------------------------------------
// Tick decoder: nested autoregressive loop, fully fused in one workgroup.
// Per tick: GRU0 (h@Whh0 WMMA + gi_c + VALU y-dot), GRU1 (fused Wih1+Whh1
// WMMA), logits (h1@t2n^T WMMA, N padded to 96), ReLU store, LDS argmax,
// d_embedding feedback.
// ---------------------------------------------------------------------------
struct TickArgs {
  const float* gi_c;      // BEAT*B x NG (c@Wih0_c^T + bih0)
  const float* ht;        // BEAT*B x 1024 (selu(b2th)) : [0:512]->h0, [512:1024]->h1
  const float* tick0;     // E
  const float* y_w;       // tick_l0 Wih f32 (NG x 522), first 10 cols = y part
  const _Float16* Whh0;   // NG x NH
  const float* bhh0;
  const _Float16* Wih1;   // NG x NH
  const _Float16* Whh1;   // NG x NH
  const float* bih1;
  const float* bhh1;
  const _Float16* t2n;    // 96 x NH (rows 90..95 zero)
  const float* t2nb;      // 90
  const float* demb;      // V x E
  float* recon;           // B x 320 x 90
};

__global__ __launch_bounds__(1024) void tick_dec(TickArgs a) {
  extern __shared__ char smem[];
  _Float16* h0b = (_Float16*)smem;                   // 2 * 32768 f16
  _Float16* h1b = h0b + 2 * NB * NH;                 // 2 * 32768 f16
  float* logits = (float*)(h1b + 2 * NB * NH);       // 64 x 96
  float* yb = logits + NB * 96;                      // 64 x 10
  const int tid = threadIdx.x;
  const int lane = tid & 31;
  const int wave = tid >> 5;

  for (int i = tid; i < NB * NE; i += 1024) yb[i] = a.tick0[i % NE];

  for (int beat = 0; beat < NBEAT; ++beat) {
    __syncthreads();
    for (int i = tid; i < NB * NH; i += 1024) {
      int b = i >> 9, c = i & (NH - 1);
      h0b[i] = (_Float16)a.ht[((size_t)beat * NB + b) * 1024 + c];
      h1b[i] = (_Float16)a.ht[((size_t)beat * NB + b) * 1024 + NH + c];
    }
    int c0 = 0, c1 = 0;
    const float* gic = a.gi_c + (size_t)beat * NB * NG;

    for (int tick = 0; tick < NTICK; ++tick) {
      __syncthreads();
      _Float16* h0c = h0b + c0 * (NB * NH);
      _Float16* h0n = h0b + (c0 ^ 1) * (NB * NH);
      _Float16* h1c = h1b + c1 * (NB * NH);
      _Float16* h1n = h1b + (c1 ^ 1) * (NB * NH);

      // ---- stage A: tick_l0 cell ----
      for (int p = 0; p < 4; ++p) {
        int pid = wave * 4 + p;
        int m0 = (pid >> 5) * 16, n0 = (pid & 31) * 16;
        v8f ar = {}, az = {}, anh = {};
        for (int k0 = 0; k0 < NH; k0 += 32) {
          v16h av = load_frag_a(h0c, NH, m0, k0, lane);
          ar  = wmma_(av, load_frag_b(a.Whh0, NH, n0,          k0, lane), ar);
          az  = wmma_(av, load_frag_b(a.Whh0, NH, n0 + NH,     k0, lane), az);
          anh = wmma_(av, load_frag_b(a.Whh0, NH, n0 + 2 * NH, k0, lane), anh);
        }
        int colj = n0 + (lane & 15);
        int rb = m0 + ((lane & 16) ? 8 : 0);
        float bhr = a.bhh0[colj], bhz = a.bhh0[NH + colj], bhn = a.bhh0[2 * NH + colj];
        for (int e = 0; e < 8; ++e) {
          int b = rb + e;
          float gr = gic[(size_t)b * NG + colj];
          float gz = gic[(size_t)b * NG + NH + colj];
          float gn = gic[(size_t)b * NG + 2 * NH + colj];
          for (int ee = 0; ee < NE; ++ee) {
            float yv = yb[b * NE + ee];
            gr += yv * a.y_w[(size_t)colj * 522 + ee];
            gz += yv * a.y_w[(size_t)(NH + colj) * 522 + ee];
            gn += yv * a.y_w[(size_t)(2 * NH + colj) * 522 + ee];
          }
          float r = sigmoidf_(gr + ar[e] + bhr);
          float z = sigmoidf_(gz + az[e] + bhz);
          float n = tanhf(gn + r * (anh[e] + bhn));
          float ho = (float)h0c[b * NH + colj];
          h0n[b * NH + colj] = (_Float16)((1.f - z) * n + z * ho);
        }
      }
      __syncthreads();

      // ---- stage B: tick_l1 cell, fused input (h0_new@Wih1) + recurrent ----
      for (int p = 0; p < 4; ++p) {
        int pid = wave * 4 + p;
        int m0 = (pid >> 5) * 16, n0 = (pid & 31) * 16;
        v8f ar = {}, az = {}, ani = {}, anh = {};
        for (int k0 = 0; k0 < NH; k0 += 32) {
          v16h ai = load_frag_a(h0n, NH, m0, k0, lane);
          ar  = wmma_(ai, load_frag_b(a.Wih1, NH, n0,          k0, lane), ar);
          az  = wmma_(ai, load_frag_b(a.Wih1, NH, n0 + NH,     k0, lane), az);
          ani = wmma_(ai, load_frag_b(a.Wih1, NH, n0 + 2 * NH, k0, lane), ani);
          v16h ah = load_frag_a(h1c, NH, m0, k0, lane);
          ar  = wmma_(ah, load_frag_b(a.Whh1, NH, n0,          k0, lane), ar);
          az  = wmma_(ah, load_frag_b(a.Whh1, NH, n0 + NH,     k0, lane), az);
          anh = wmma_(ah, load_frag_b(a.Whh1, NH, n0 + 2 * NH, k0, lane), anh);
        }
        int colj = n0 + (lane & 15);
        int rb = m0 + ((lane & 16) ? 8 : 0);
        float bir = a.bih1[colj] + a.bhh1[colj];
        float biz = a.bih1[NH + colj] + a.bhh1[NH + colj];
        float bin_ = a.bih1[2 * NH + colj];
        float bhn = a.bhh1[2 * NH + colj];
        for (int e = 0; e < 8; ++e) {
          int b = rb + e;
          float r = sigmoidf_(ar[e] + bir);
          float z = sigmoidf_(az[e] + biz);
          float n = tanhf(ani[e] + bin_ + r * (anh[e] + bhn));
          float ho = (float)h1c[b * NH + colj];
          h1n[b * NH + colj] = (_Float16)((1.f - z) * n + z * ho);
        }
      }
      __syncthreads();

      // ---- stage C: logits = relu(h1_new @ t2n^T + b), write recon ----
      if (wave < 24) {
        int m0 = (wave / 6) * 16, n0 = (wave % 6) * 16;
        v8f acc = {};
        for (int k0 = 0; k0 < NH; k0 += 32)
          acc = wmma_(load_frag_a(h1n, NH, m0, k0, lane),
                      load_frag_b(a.t2n, NH, n0, k0, lane), acc);
        int col = n0 + (lane & 15);
        int rb = m0 + ((lane & 16) ? 8 : 0);
        float bv = (col < NV) ? a.t2nb[col] : 0.f;
        int seq = beat * NTICK + tick;
        for (int e = 0; e < 8; ++e) {
          int b = rb + e;
          float v = fmaxf(acc[e] + bv, 0.f);
          logits[b * 96 + col] = (col < NV) ? v : -1.f;
          if (col < NV)
            a.recon[(size_t)b * (NT * NV) + (size_t)seq * NV + col] = v;
        }
      }
      __syncthreads();

      // ---- stage D: argmax feedback (first-max, like jnp.argmax) ----
      if (tid < NB) {
        int best = 0;
        float bm = logits[tid * 96];
        for (int v = 1; v < NV; ++v) {
          float lv = logits[tid * 96 + v];
          if (lv > bm) { bm = lv; best = v; }
        }
        for (int e = 0; e < NE; ++e) yb[tid * NE + e] = a.demb[best * NE + e];
      }
      c0 ^= 1;
      c1 ^= 1;
    }
  }
}

// ---------------------------------------------------------------------------
// Host orchestration
// ---------------------------------------------------------------------------
extern "C" void kernel_launch(void* const* d_in, const int* in_sizes, int n_in,
                              void* d_out, int out_size, void* d_ws, size_t ws_size,
                              hipStream_t stream) {
  (void)in_sizes; (void)n_in; (void)out_size; (void)ws_size;
  auto F = [&](int i) { return (const float*)d_in[i]; };
  const int* x = (const int*)d_in[0];
  const float* stdz = F(2);
  const float* emb = F(3);
  const float* demb = F(4);

  // workspace allocator
  char* ws = (char*)d_ws;
  size_t off = 0;
  auto alloc = [&](size_t bytes) -> char* {
    off = (off + 255) & ~(size_t)255;
    char* p = ws + off;
    off += bytes;
    return p;
  };
  auto allocH = [&](size_t n) { return (_Float16*)alloc(n * 2); };
  auto allocF = [&](size_t n) { return (float*)alloc(n * 4); };

  // f16 weights
  _Float16* WhhL0f = allocH((size_t)NG * NH);
  _Float16* WhhL0b = allocH((size_t)NG * NH);
  _Float16* WihL1f = allocH((size_t)NG * 1024);
  _Float16* WhhL1f = allocH((size_t)NG * NH);
  _Float16* WihL1b = allocH((size_t)NG * 1024);
  _Float16* WhhL1b = allocH((size_t)NG * NH);
  _Float16* muW16  = allocH((size_t)NZ * 2048);
  _Float16* lvW16  = allocH((size_t)NZ * 2048);
  _Float16* z2bhW16 = allocH((size_t)1024 * NZ);
  _Float16* WhhB0  = allocH((size_t)NG * NH);
  _Float16* WihB1  = allocH((size_t)NG * NH);
  _Float16* WhhB1  = allocH((size_t)NG * NH);
  _Float16* b2thW16 = allocH((size_t)1024 * NH);
  _Float16* b2tiW16 = allocH((size_t)NH * NH);
  _Float16* WihcT0 = allocH((size_t)NG * NH);  // cols 10..521 of tick_l0 Wih
  _Float16* WhhT0  = allocH((size_t)NG * NH);
  _Float16* WihT1  = allocH((size_t)NG * NH);
  _Float16* WhhT1  = allocH((size_t)NG * NH);
  _Float16* t2n96  = allocH((size_t)96 * NH);
  // activations
  _Float16* out0_16 = allocH((size_t)NT * NB * 1024);   // encoder L0 bi output
  _Float16* hcat16  = allocH((size_t)NB * 2048);
  float*    lvf32   = allocF((size_t)NB * NZ);
  _Float16* z16     = allocH((size_t)NB * NZ);
  float*    hbt     = allocF((size_t)NB * 1024);        // selu(z2bh)
  _Float16* bo0_16  = allocH((size_t)NBEAT * NB * NH);
  _Float16* beat16  = allocH((size_t)NBEAT * NB * NH);
  float*    htdec   = allocF((size_t)NBEAT * NB * 1024);  // selu(b2th)
  _Float16* c16     = allocH((size_t)NBEAT * NB * NH);    // selu(b2ti) f16
  float*    gi_c    = allocF((size_t)NBEAT * NB * NG);
  float*    gib0    = allocF(NG);

  // d_out layout: recon | z | mu | var
  float* out_recon = (float*)d_out;
  float* out_z   = out_recon + (size_t)NB * NT * NV;
  float* out_mu  = out_z + (size_t)NB * NZ;
  float* out_var = out_mu + (size_t)NB * NZ;

  // ---- weight conversions ----
  auto cvt = [&](const float* src, _Float16* dst, int rows, int cols,
                 int sstride, int soff, int srows) {
    int n = rows * cols;
    cvt_f16<<<(n + 255) / 256, 256, 0, stream>>>(src, dst, rows, cols, sstride,
                                                 soff, cols, srows);
  };
  cvt(F(6),  WhhL0f, NG, NH, NH, 0, NG);
  cvt(F(10), WhhL0b, NG, NH, NH, 0, NG);
  cvt(F(13), WihL1f, NG, 1024, 1024, 0, NG);
  cvt(F(14), WhhL1f, NG, NH, NH, 0, NG);
  cvt(F(17), WihL1b, NG, 1024, 1024, 0, NG);
  cvt(F(18), WhhL1b, NG, NH, NH, 0, NG);
  cvt(F(21), muW16, NZ, 2048, 2048, 0, NZ);
  cvt(F(23), lvW16, NZ, 2048, 2048, 0, NZ);
  cvt(F(25), z2bhW16, 1024, NZ, NZ, 0, 1024);
  cvt(F(29), WhhB0, NG, NH, NH, 0, NG);
  cvt(F(32), WihB1, NG, NH, NH, 0, NG);
  cvt(F(33), WhhB1, NG, NH, NH, 0, NG);
  cvt(F(36), b2thW16, 1024, NH, NH, 0, 1024);
  cvt(F(38), b2tiW16, NH, NH, NH, 0, NH);
  cvt(F(41), WihcT0, NG, NH, 522, 10, NG);   // repack c-part of tick_l0 Wih
  cvt(F(42), WhhT0, NG, NH, NH, 0, NG);
  cvt(F(45), WihT1, NG, NH, NH, 0, NG);
  cvt(F(46), WhhT1, NG, NH, NH, 0, NG);
  cvt(F(49), t2n96, 96, NH, NH, 0, NV);      // zero-pad rows 90..95
  gib0_k<<<(NG + 255) / 256, 256, 0, stream>>>(F(27), F(28), F(30), gib0);

  const size_t SMEM_SEQ = (size_t)2 * NB * NH * 2 + (size_t)NB * 16 * 4;
  auto gemm = [&](const _Float16* A, int lda, const _Float16* W, int ldw,
                  const float* bias, float* Cf, _Float16* Ch, int ldc, int M,
                  int N, int K, int act) {
    int tiles = (M / 16) * (N / 16);
    gemm_wmma<<<(tiles + 7) / 8, 256, 0, stream>>>(A, lda, W, ldw, bias, Cf, Ch,
                                                   ldc, M, N, K, act);
  };

  // ---- encoder L0 (bidirectional, 2 persistent blocks) ----
  GruJob jf = {};
  jf.Whh = WhhL0f; jf.bhh = F(8); jf.bih = F(7);
  jf.mode = 0; jf.xtok = x; jf.emb = emb; jf.WihF32 = F(5); jf.E_ = NE;
  jf.hout_f16 = out0_16; jf.out_stride = 1024;
  jf.hfinal_f16 = hcat16; jf.fin_stride = 2048;
  jf.T = NT; jf.reverse = 0;
  GruJob jb = jf;
  jb.Whh = WhhL0b; jb.bhh = F(12); jb.bih = F(11); jb.WihF32 = F(9);
  jb.hout_f16 = out0_16 + NH; jb.hfinal_f16 = hcat16 + NH; jb.reverse = 1;
  gru_seq<<<2, 1024, SMEM_SEQ, stream>>>(jf, jb);

  // ---- encoder L1 (bidirectional, fused input GEMM; only final h kept) ----
  GruJob k1 = {};
  k1.mode = 1; k1.xs = out0_16; k1.Din = 1024;
  k1.Wih = WihL1f; k1.bih = F(15); k1.Whh = WhhL1f; k1.bhh = F(16);
  k1.hfinal_f16 = hcat16 + 1024; k1.fin_stride = 2048;
  k1.T = NT; k1.reverse = 0;
  GruJob k2 = k1;
  k2.Wih = WihL1b; k2.bih = F(19); k2.Whh = WhhL1b; k2.bhh = F(20);
  k2.hfinal_f16 = hcat16 + 1536; k2.reverse = 1;
  gru_seq<<<2, 1024, SMEM_SEQ, stream>>>(k1, k2);

  // ---- latent ----
  gemm(hcat16, 2048, muW16, 2048, F(22), out_mu, nullptr, NZ, NB, NZ, 2048, 0);
  gemm(hcat16, 2048, lvW16, 2048, F(24), lvf32, nullptr, NZ, NB, NZ, 2048, 0);
  reparam_k<<<(NB * NZ + 255) / 256, 256, 0, stream>>>(out_mu, lvf32, stdz,
                                                       out_z, out_var, z16,
                                                       NB * NZ);
  gemm(z16, NZ, z2bhW16, NZ, F(26), hbt, nullptr, 1024, NB, 1024, NZ, 1);

  // ---- beat decoder ----
  GruJob b0 = {};
  b0.mode = 2; b0.givec = gib0; b0.Whh = WhhB0; b0.bhh = F(31);
  b0.h0 = hbt; b0.h0_stride = 1024; b0.T = NBEAT;
  b0.hout_f16 = bo0_16; b0.out_stride = NH;
  gru_seq<<<1, 1024, SMEM_SEQ, stream>>>(b0, b0);

  GruJob b1 = {};
  b1.mode = 1; b1.xs = bo0_16; b1.Din = NH;
  b1.Wih = WihB1; b1.bih = F(34); b1.Whh = WhhB1; b1.bhh = F(35);
  b1.h0 = hbt + NH; b1.h0_stride = 1024; b1.T = NBEAT;
  b1.hout_f16 = beat16; b1.out_stride = NH;
  gru_seq<<<1, 1024, SMEM_SEQ, stream>>>(b1, b1);

  // ---- per-beat projections for tick decoder ----
  gemm(beat16, NH, b2thW16, NH, F(37), htdec, nullptr, 1024, NBEAT * NB, 1024, NH, 1);
  gemm(beat16, NH, b2tiW16, NH, F(39), nullptr, c16, NH, NBEAT * NB, NH, NH, 1);
  gemm(c16, NH, WihcT0, NH, F(43), gi_c, nullptr, NG, NBEAT * NB, NG, NH, 0);

  // ---- tick decoder ----
  TickArgs ta = {};
  ta.gi_c = gi_c; ta.ht = htdec; ta.tick0 = F(40); ta.y_w = F(41);
  ta.Whh0 = WhhT0; ta.bhh0 = F(44);
  ta.Wih1 = WihT1; ta.Whh1 = WhhT1; ta.bih1 = F(47); ta.bhh1 = F(48);
  ta.t2n = t2n96; ta.t2nb = F(50); ta.demb = demb; ta.recon = out_recon;
  const size_t SMEM_TICK =
      (size_t)4 * NB * NH * 2 + (size_t)NB * 96 * 4 + (size_t)NB * NE * 4;
  tick_dec<<<1, 1024, SMEM_TICK, stream>>>(ta);
}
